// ClusteringLayer_35064113004990
// MI455X (gfx1250) — matrix operand order repaired and checked by reading, not compiled, not run on hardware
//
#include <hip/hip_runtime.h>

typedef __bf16 bf16;
typedef __attribute__((ext_vector_type(16))) __bf16 v16bf;
typedef __attribute__((ext_vector_type(8)))  __bf16 v8bf;
typedef __attribute__((ext_vector_type(8)))  float  v8f;
typedef __attribute__((ext_vector_type(4)))  float  v4f;
typedef __attribute__((ext_vector_type(4)))  unsigned int v4u;
typedef __attribute__((ext_vector_type(4)))  int v4i;

#define DDIM     512            // feature dim (K of the GEMM)
#define NCLUST   256            // clusters (N of the GEMM)
#define M_TILE   64             // rows per block
#define KCH      32             // K per WMMA chunk
#define NKCH     (DDIM / KCH)   // 16 chunks
#define A_STRIDE 520            // padded bf16 row stride for A in LDS
#define B_STRIDE 272            // padded bf16 row stride for B chunk in LDS
#define B_BUF_ELEMS (2 * KCH * B_STRIDE)   // one double-buffer slot (hi+lo)

// ---- CDNA5 async global->LDS path (ASYNCcnt), with safe fallback ----------
#if defined(__has_builtin)
# if __has_builtin(__builtin_amdgcn_global_load_async_to_lds_b128)
#  define ASYNC_LDS 1
# endif
#endif
#ifndef ASYNC_LDS
# define ASYNC_LDS 0
#endif

__device__ __forceinline__ void copy16_to_lds(const bf16* g, bf16* l) {
#if ASYNC_LDS
    // param 1 is addrspace(1) ("__device__") -> (global_src, lds_dst, off, cpol)
    __builtin_amdgcn_global_load_async_to_lds_b128((v4i*)g, (v4i*)l, 0, 0);
#else
    *(v4u*)l = *(const v4u*)g;
#endif
}

__device__ __forceinline__ void wait_async_lds() {
#if ASYNC_LDS
# if defined(__has_builtin) && __has_builtin(__builtin_amdgcn_s_wait_asynccnt)
    __builtin_amdgcn_s_wait_asynccnt(0);
# else
    asm volatile("s_wait_asynccnt 0x0" ::: "memory");
# endif
#endif
}

// ---------------------------------------------------------------------------
// Prep: clusters (256x512 f32) -> K-major bf16 hi/lo Bt[k][n] + c2[n].
// ---------------------------------------------------------------------------
__global__ __launch_bounds__(512) void prep_clusters(
    const float* __restrict__ clusters,
    bf16* __restrict__ bt_hi, bf16* __restrict__ bt_lo,
    float* __restrict__ c2)
{
    __shared__ float red[512];
    const int n = blockIdx.x;
    const int k = threadIdx.x;
    float v = clusters[n * DDIM + k];
    bf16 h = (bf16)v;
    bf16 l = (bf16)(v - (float)h);
    bt_hi[k * NCLUST + n] = h;
    bt_lo[k * NCLUST + n] = l;
    red[k] = v * v;
    __syncthreads();
    for (int s = 256; s > 0; s >>= 1) {
        if (k < s) red[k] += red[k + s];
        __syncthreads();
    }
    if (k == 0) c2[n] = red[0];
}

// ---------------------------------------------------------------------------
// Main: fused (x @ clusters^T) -> 1/(1+d^2) -> row-normalize.
// 1024 blocks x 256 threads (8 waves). Wave w: M-subtile (w&3), N-half (w>>2).
// B chunks double-buffered in LDS, filled with async global->LDS loads.
// ---------------------------------------------------------------------------
extern __shared__ char smem[];

__device__ __forceinline__ void stage_b_chunk(
    const bf16* __restrict__ bt_hi, const bf16* __restrict__ bt_lo,
    int kb, int tid, bf16* dh)                 // dl = dh + KCH*B_STRIDE
{
#pragma unroll
    for (int j = 0; j < 4; ++j) {
        int i  = tid * 4 + j;                  // 1024 x 16B per array
        int kr = i >> 5;
        int nc = (i & 31) << 3;
        copy16_to_lds(bt_hi + (kb + kr) * NCLUST + nc, dh + kr * B_STRIDE + nc);
        copy16_to_lds(bt_lo + (kb + kr) * NCLUST + nc,
                      dh + KCH * B_STRIDE + kr * B_STRIDE + nc);
    }
}

__global__ __launch_bounds__(256) void dec_soft_assign(
    const float* __restrict__ x,
    const bf16*  __restrict__ bt_hi,
    const bf16*  __restrict__ bt_lo,
    const float* __restrict__ c2g,
    float*       __restrict__ out)
{
    // dynamic LDS layout (single base pointer; buffers selected by offset
    // arithmetic so InferAddressSpaces keeps everything in addrspace(3))
    bf16* Ahi   = (bf16*)smem;                      // [64][520]
    bf16* Alo   = Ahi + M_TILE * A_STRIDE;          // [64][520]
    bf16* Bbase = Alo + M_TILE * A_STRIDE;          // 2 x {hi[32][272], lo[32][272]}

    __shared__ float x2s[M_TILE];
    __shared__ float rowsum[2][M_TILE];
    __shared__ float c2s[NCLUST];

    const int tid  = threadIdx.x;
    const int wave = tid >> 5;
    const int lane = tid & 31;
    const long mbase = (long)blockIdx.x * M_TILE;

    if (tid < M_TILE) x2s[tid] = 0.f;
    c2s[tid] = c2g[tid];                            // 256 threads == NCLUST
    __syncthreads();

    // prologue: kick off chunk 0 into buffer 0 (async overlaps with A staging)
    stage_b_chunk(bt_hi, bt_lo, 0, tid, Bbase);

    // ---- stage x tile (64x512 f32) as bf16 hi/lo into LDS; row norms ----
    {
        const int row   = tid >> 2;                 // 4 threads per row
        const int cbase = (tid & 3) * 128;          // 128 contiguous floats each
        const float* xr = x + (mbase + row) * DDIM + cbase;
        bf16* ah = Ahi + row * A_STRIDE + cbase;
        bf16* al = Alo + row * A_STRIDE + cbase;
        float ss = 0.f;
#pragma unroll
        for (int i = 0; i < 16; ++i) {              // 16 * 8 floats
            v4f f0 = *(const v4f*)(xr + i * 8);
            v4f f1 = *(const v4f*)(xr + i * 8 + 4);
            v8bf h, l;
#pragma unroll
            for (int j = 0; j < 4; ++j) {
                float a = f0[j], b = f1[j];
                ss += a * a + b * b;
                bf16 ha = (bf16)a; h[j]     = ha; l[j]     = (bf16)(a - (float)ha);
                bf16 hb = (bf16)b; h[4 + j] = hb; l[4 + j] = (bf16)(b - (float)hb);
            }
            *(v8bf*)(ah + i * 8) = h;
            *(v8bf*)(al + i * 8) = l;
        }
        atomicAdd(&x2s[row], ss);                   // ds_add_f32
    }

    const int mtile = wave & 3;
    const int half  = wave >> 2;

    v8f acc[8];
#pragma unroll
    for (int i = 0; i < 8; ++i) acc[i] = (v8f)(0.f);

    // per-lane A fragment base: lanes 0-15 -> K [0..8)+[16..24), lanes 16-31 -> +8
    const int am  = mtile * 16 + (lane & 15);
    const int ak0 = (lane >> 4) * 8;
    const bf16* aph = Ahi + am * A_STRIDE + ak0;
    const bf16* apl = Alo + am * A_STRIDE + ak0;

    const int bn = half * 128;                      // wave's N base

    for (int kc = 0; kc < NKCH; ++kc) {
        const int kb  = kc * KCH;
        const int cur = kc & 1;

        wait_async_lds();                           // own chunk-kc asyncs done
        __syncthreads();                            // all waves' chunk kc visible;
                                                    // all reads of other buffer done
        if (kc + 1 < NKCH)                          // prefetch next chunk, other buffer
            stage_b_chunk(bt_hi, bt_lo, kb + KCH, tid,
                          Bbase + (cur ^ 1) * B_BUF_ELEMS);

        // ---- A fragments (hi/lo), reused across 8 N-tiles ----
        v16bf a_hi, a_lo;
        ((v8bf*)&a_hi)[0] = *(const v8bf*)(aph + kb);
        ((v8bf*)&a_hi)[1] = *(const v8bf*)(aph + kb + 16);
        ((v8bf*)&a_lo)[0] = *(const v8bf*)(apl + kb);
        ((v8bf*)&a_lo)[1] = *(const v8bf*)(apl + kb + 16);

        // B fragments from current buffer (pure offset arithmetic -> ds_load)
        const bf16* bp_h = Bbase + cur * B_BUF_ELEMS + (lane >> 4) * 8 * B_STRIDE + bn + (lane & 15);
        const bf16* bp_l = bp_h + KCH * B_STRIDE;
#pragma unroll
        for (int nt = 0; nt < 8; ++nt) {
            v16bf b_hi, b_lo;
#pragma unroll
            for (int s = 0; s < 16; ++s) {
                const int kk = (s & 7) + ((s >> 3) << 4);
                b_hi[s] = bp_h[kk * B_STRIDE + nt * 16];
                b_lo[s] = bp_l[kk * B_STRIDE + nt * 16];
            }
            acc[nt] = __builtin_amdgcn_wmma_f32_16x16x32_bf16(false, a_hi, false, b_hi,
                                                              (short)0, acc[nt], false, false);
            acc[nt] = __builtin_amdgcn_wmma_f32_16x16x32_bf16(false, a_hi, false, b_lo,
                                                              (short)0, acc[nt], false, false);
            acc[nt] = __builtin_amdgcn_wmma_f32_16x16x32_bf16(false, a_lo, false, b_hi,
                                                              (short)0, acc[nt], false, false);
        }
    }

    // ---- epilogue: q = 1/(1+d^2), row sums, normalize ----
    const int m0   = mtile * 16 + (lane >> 4) * 8;  // C layout: VGPR r -> row m0+r
    const int ncol = bn + (lane & 15);

    float rp[8];
#pragma unroll
    for (int r = 0; r < 8; ++r) rp[r] = 0.f;

#pragma unroll
    for (int nt = 0; nt < 8; ++nt) {
        float cc = c2s[ncol + nt * 16];
#pragma unroll
        for (int r = 0; r < 8; ++r) {
            float d2 = fmaxf(x2s[m0 + r] + cc - 2.f * acc[nt][r], 0.f);
            float qv = 1.0f / (1.0f + d2);          // ALPHA=1 -> exponent is 1
            acc[nt][r] = qv;
            rp[r] += qv;
        }
    }
    // reduce across the 16 lanes holding one row (xor masks stay within a half)
#pragma unroll
    for (int r = 0; r < 8; ++r) {
#pragma unroll
        for (int m = 1; m <= 8; m <<= 1) rp[r] += __shfl_xor(rp[r], m, 32);
    }
    if ((lane & 15) == 0) {
#pragma unroll
        for (int r = 0; r < 8; ++r) rowsum[half][m0 + r] = rp[r];
    }
    __syncthreads();

#pragma unroll
    for (int r = 0; r < 8; ++r) {
        float invs = 1.0f / (rowsum[0][m0 + r] + rowsum[1][m0 + r]);
        float* orow = out + (mbase + m0 + r) * NCLUST + ncol;
#pragma unroll
        for (int nt = 0; nt < 8; ++nt) orow[nt * 16] = acc[nt][r] * invs;
    }
}

// ---------------------------------------------------------------------------
extern "C" void kernel_launch(void* const* d_in, const int* in_sizes, int n_in,
                              void* d_out, int out_size, void* d_ws, size_t ws_size,
                              hipStream_t stream) {
    const float* x        = (const float*)d_in[0];   // (65536, 512)
    const float* clusters = (const float*)d_in[1];   // (256, 512)
    float* out            = (float*)d_out;           // (65536, 256)

    // workspace layout: c2 (1 KB) | Bt_hi (256 KB) | Bt_lo (256 KB)
    float* c2  = (float*)d_ws;
    bf16* bthi = (bf16*)((char*)d_ws + 1024);
    bf16* btlo = bthi + (size_t)DDIM * NCLUST;

    prep_clusters<<<NCLUST, 512, 0, stream>>>(clusters, bthi, btlo, c2);

    const int rows = in_sizes[0] / DDIM;             // 65536
    const int nblk = rows / M_TILE;                  // 1024
    const size_t lds = (size_t)(2 * M_TILE * A_STRIDE + 2 * B_BUF_ELEMS) * sizeof(bf16);

    (void)hipFuncSetAttribute((const void*)dec_soft_assign,
                              hipFuncAttributeMaxDynamicSharedMemorySize, (int)lds);
    dec_soft_assign<<<nblk, 256, lds, stream>>>(x, bthi, btlo, c2, out);
}
